// PNA_6012954214820
// MI455X (gfx1250) — matrix operand order repaired and checked by reading, not compiled, not run on hardware
//
#include <hip/hip_runtime.h>

#define NN 50000
#define NE 800000
#define FD 64
#define NL 3
#define EPSV 1e-5f

typedef __attribute__((ext_vector_type(16))) _Float16 v16h;
typedef __attribute__((ext_vector_type(8)))  float    v8f;

__device__ __forceinline__ void atomicMaxF(float* a, float v) {
  if (v >= 0.0f) atomicMax((int*)a, __float_as_int(v));
  else           atomicMin((unsigned int*)a, __float_as_uint(v));
}
__device__ __forceinline__ void atomicMinF(float* a, float v) {
  if (v >= 0.0f) atomicMin((int*)a, __float_as_int(v));
  else           atomicMax((unsigned int*)a, __float_as_uint(v));
}

union H8U { uint4 u; _Float16 h[8]; };

// ---------------- small prep kernels ----------------
__global__ void k_zero(float* deg, float* sumlog) {
  int i = blockIdx.x * blockDim.x + threadIdx.x;
  if (i < NN) deg[i] = 0.0f;
  if (i == 0) sumlog[0] = 0.0f;
}
__global__ void k_deg(const int* __restrict__ dstp, float* deg) {
  int i = blockIdx.x * blockDim.x + threadIdx.x;
  if (i < NE) atomicAdd(deg + dstp[i], 1.0f);
}
__global__ void k_logdeg(const float* __restrict__ deg, float* logdeg, float* sumlog) {
  int i = blockIdx.x * blockDim.x + threadIdx.x;
  if (i < NN) {
    float dc = fmaxf(deg[i], 1.0f);
    float ld = logf(dc + 1.0f);
    logdeg[i] = ld;
    atomicAdd(sumlog, ld);
  }
}
__global__ void k_scalers(const float* __restrict__ logdeg, const float* __restrict__ sumlog,
                          float* amp, float* att) {
  int i = blockIdx.x * blockDim.x + threadIdx.x;
  if (i < NN) {
    float avg = sumlog[0] / (float)NN;
    float ld = logdeg[i];
    amp[i] = ld / avg;
    att[i] = avg / ld;
  }
}
// convert weights fp32 [L][K][N] -> f16 transposed [L][N][K]
__global__ void k_wcvt(const float* __restrict__ src, _Float16* dst, int K, int Nc, int total) {
  int i = blockIdx.x * blockDim.x + threadIdx.x;
  if (i < total) {
    int per = K * Nc;
    int l = i / per;
    int rem = i - l * per;
    int k = rem / Nc;
    int n = rem - k * Nc;
    dst[l * per + n * K + k] = (_Float16)src[i];
  }
}
__global__ void k_fill(float* s1, float* s2, float* mx, float* mn) {
  long i = (long)blockIdx.x * blockDim.x + threadIdx.x;
  if (i < (long)NN * FD) {
    s1[i] = 0.0f; s2[i] = 0.0f;
    mx[i] = -__builtin_inff();
    mn[i] =  __builtin_inff();
  }
}

// ---------------- edge kernel: fused pre-GEMM (WMMA) + multi-agg scatter ----------------
// one wave32 computes 16 edges x 64 features, pre_wT (f16 [64][128]) staged in LDS
__global__ void k_edge(const float* __restrict__ h,
                       const int* __restrict__ srcp, const int* __restrict__ dstp,
                       const _Float16* __restrict__ preT, const float* __restrict__ preb,
                       float* s1, float* s2, float* mx, float* mn) {
  __shared__ _Float16 w[64 * 128];
  { // 16 KB cooperative load
    const uint4* g = (const uint4*)preT;
    uint4* l = (uint4*)w;
    for (int i = threadIdx.x; i < 64 * 128 * 2 / 16; i += blockDim.x) l[i] = g[i];
  }
  __syncthreads();

  int wave = threadIdx.x >> 5;
  int lane = threadIdx.x & 31;
  int lr = lane & 15, half = lane >> 4;
  int tile = blockIdx.x * (blockDim.x >> 5) + wave;
  int base = tile * 16;
  if (base >= NE) return; // uniform per wave

  int e = base + lr;
  const float* rowD = h + (long)dstp[e] * FD;
  const float* rowS = h + (long)srcp[e] * FD;

  v8f c[4] = {};
#pragma unroll
  for (int ch = 0; ch < 4; ++ch) {
    v16h a;
#pragma unroll
    for (int piece = 0; piece < 2; ++piece) {
      int k0 = ch * 32 + (half ? 16 : 0) + piece * 8;
      const float* r = (k0 < 64) ? (rowD + k0) : (rowS + (k0 - 64));
      float4 f0 = ((const float4*)r)[0];
      float4 f1 = ((const float4*)r)[1];
      int o = piece * 8;
      a[o + 0] = (_Float16)f0.x; a[o + 1] = (_Float16)f0.y;
      a[o + 2] = (_Float16)f0.z; a[o + 3] = (_Float16)f0.w;
      a[o + 4] = (_Float16)f1.x; a[o + 5] = (_Float16)f1.y;
      a[o + 6] = (_Float16)f1.z; a[o + 7] = (_Float16)f1.w;
    }
    // preload all four B fragments of this chunk, then issue WMMAs back-to-back
    v16h btile[4];
#pragma unroll
    for (int t = 0; t < 4; ++t)
      btile[t] = *(const v16h*)&w[(t * 16 + lr) * 128 + ch * 32 + (half ? 16 : 0)];
#pragma unroll
    for (int t = 0; t < 4; ++t)
      c[t] = __builtin_amdgcn_wmma_f32_16x16x32_f16(false, a, false, btile[t], (short)0, c[t], false, false);
  }
  // dst nodes for the 8 C rows this lane holds (M = r + 8*half)
  int dn[8];
#pragma unroll
  for (int r = 0; r < 8; ++r) dn[r] = dstp[base + half * 8 + r];
#pragma unroll
  for (int t = 0; t < 4; ++t) {
    float bias = preb[t * 16 + lr];
#pragma unroll
    for (int r = 0; r < 8; ++r) {
      float v = c[t][r] + bias;
      long addr = (long)dn[r] * FD + t * 16 + lr;
      atomicAdd(s1 + addr, v);
      atomicAdd(s2 + addr, v * v);
      atomicMaxF(mx + addr, v);
      atomicMinF(mn + addr, v);
    }
  }
}

// ---------------- aggregator finalize: mean/max/min/std -> agg[N][256] ----------------
__global__ void k_agg(const float* __restrict__ deg, const float* __restrict__ s1,
                      const float* __restrict__ s2, const float* __restrict__ mx,
                      const float* __restrict__ mn, float* agg) {
  long i = (long)blockIdx.x * blockDim.x + threadIdx.x;
  if (i >= (long)NN * FD) return;
  int node = (int)(i >> 6);
  int f = (int)(i & 63);
  float d = deg[node];
  float dc = fmaxf(d, 1.0f);
  float mean = s1[i] / dc;
  float var = s2[i] / dc - mean * mean;
  float sd = sqrtf(fmaxf(var, 0.0f) + EPSV);
  bool he = d > 0.0f;
  long b = (long)node * 256 + f;
  agg[b]       = mean;
  agg[b + 64]  = he ? mx[i] : 0.0f;
  agg[b + 128] = he ? mn[i] : 0.0f;
  agg[b + 192] = sd;
}

// ---------------- node kernel: post-GEMM (K=832) + lin-GEMM (K=64) + residual ----------------
// 25 waves/WG (3125 tiles = 25*125). A built on the fly: k<64 -> h, else agg * {1,amp,att}.
__global__ void k_node(const float* __restrict__ hprev, const float* __restrict__ agg,
                       const float* __restrict__ amp, const float* __restrict__ att,
                       const _Float16* __restrict__ postT, const float* __restrict__ postb,
                       const _Float16* __restrict__ linT, const float* __restrict__ linb,
                       float* hnext) {
  extern __shared__ char smem[];
  _Float16* wpost = (_Float16*)smem;          // 64*832
  _Float16* wlin  = wpost + 64 * 832;         // 64*64
  _Float16* stage = wlin + 64 * 64;           // nwaves * 16*64
  {
    const uint4* g0 = (const uint4*)postT;
    uint4* l0 = (uint4*)wpost;
    for (int i = threadIdx.x; i < 64 * 832 * 2 / 16; i += blockDim.x) l0[i] = g0[i];
    const uint4* g1 = (const uint4*)linT;
    uint4* l1 = (uint4*)wlin;
    for (int i = threadIdx.x; i < 64 * 64 * 2 / 16; i += blockDim.x) l1[i] = g1[i];
  }
  __syncthreads();

  int wave = threadIdx.x >> 5;
  int lane = threadIdx.x & 31;
  int lr = lane & 15, half = lane >> 4;
  int tile = blockIdx.x * (blockDim.x >> 5) + wave;
  int base = tile * 16;
  int m = base + lr;
  float ampv = amp[m], attv = att[m];
  const float* hr = hprev + (long)m * FD;
  const float* ar = agg + (long)m * 256;

  v8f c[4] = {};
#pragma unroll 2
  for (int ch = 0; ch < 26; ++ch) {
    v16h a;
#pragma unroll
    for (int piece = 0; piece < 2; ++piece) {
      int k0 = ch * 32 + (half ? 16 : 0) + piece * 8;
      const float* r;
      float sc;
      if (k0 < 64) { r = hr + k0; sc = 1.0f; }
      else {
        int kp = k0 - 64;
        int sel = kp >> 8;
        r = ar + (kp & 255);
        sc = (sel == 0) ? 1.0f : ((sel == 1) ? ampv : attv);
      }
      float4 f0 = ((const float4*)r)[0];
      float4 f1 = ((const float4*)r)[1];
      int o = piece * 8;
      a[o + 0] = (_Float16)(f0.x * sc); a[o + 1] = (_Float16)(f0.y * sc);
      a[o + 2] = (_Float16)(f0.z * sc); a[o + 3] = (_Float16)(f0.w * sc);
      a[o + 4] = (_Float16)(f1.x * sc); a[o + 5] = (_Float16)(f1.y * sc);
      a[o + 6] = (_Float16)(f1.z * sc); a[o + 7] = (_Float16)(f1.w * sc);
    }
    v16h btile[4];
#pragma unroll
    for (int t = 0; t < 4; ++t)
      btile[t] = *(const v16h*)&wpost[(t * 16 + lr) * 832 + ch * 32 + (half ? 16 : 0)];
#pragma unroll
    for (int t = 0; t < 4; ++t)
      c[t] = __builtin_amdgcn_wmma_f32_16x16x32_f16(false, a, false, btile[t], (short)0, c[t], false, false);
  }

  // bias, stage out tile (16x64) as f16 in LDS for the lin GEMM
  _Float16* st = stage + wave * 16 * 64;
#pragma unroll
  for (int t = 0; t < 4; ++t) {
    float bias = postb[t * 16 + lr];
#pragma unroll
    for (int r = 0; r < 8; ++r)
      st[(r + half * 8) * 64 + t * 16 + lr] = (_Float16)(c[t][r] + bias);
  }
  __syncthreads();

  v8f c2[4] = {};
#pragma unroll
  for (int ch = 0; ch < 2; ++ch) {
    v16h a;
    const _Float16* rowp = st + lr * 64;
#pragma unroll
    for (int piece = 0; piece < 2; ++piece) {
      int k0 = ch * 32 + (half ? 16 : 0) + piece * 8;
      H8U u;
      u.u = *(const uint4*)(rowp + k0);
      int o = piece * 8;
#pragma unroll
      for (int j = 0; j < 8; ++j) a[o + j] = u.h[j];
    }
    v16h btile[4];
#pragma unroll
    for (int t = 0; t < 4; ++t)
      btile[t] = *(const v16h*)&wlin[(t * 16 + lr) * 64 + ch * 32 + (half ? 16 : 0)];
#pragma unroll
    for (int t = 0; t < 4; ++t)
      c2[t] = __builtin_amdgcn_wmma_f32_16x16x32_f16(false, a, false, btile[t], (short)0, c2[t], false, false);
  }

#pragma unroll
  for (int t = 0; t < 4; ++t) {
    float bias = linb[t * 16 + lr];
#pragma unroll
    for (int r = 0; r < 8; ++r) {
      int row = base + r + half * 8;
      long addr = (long)row * FD + t * 16 + lr;
      hnext[addr] = c2[t][r] + bias + hprev[addr];
    }
  }
}

// ---------------- host orchestration ----------------
extern "C" void kernel_launch(void* const* d_in, const int* in_sizes, int n_in,
                              void* d_out, int out_size, void* d_ws, size_t ws_size,
                              hipStream_t stream) {
  const float* x      = (const float*)d_in[0];
  const int*   ei     = (const int*)d_in[1];
  const float* pre_w  = (const float*)d_in[2];
  const float* pre_b  = (const float*)d_in[3];
  const float* post_w = (const float*)d_in[4];
  const float* post_b = (const float*)d_in[5];
  const float* lin_w  = (const float*)d_in[6];
  const float* lin_b  = (const float*)d_in[7];
  const int* srcp = ei;
  const int* dstp = ei + NE;

  char* p = (char*)d_ws;
  auto alloc = [&](size_t bytes) { char* r = p; p += (bytes + 255) & ~(size_t)255; return r; };
  float* deg    = (float*)alloc((size_t)NN * 4);
  float* logdeg = (float*)alloc((size_t)NN * 4);
  float* amp    = (float*)alloc((size_t)NN * 4);
  float* att    = (float*)alloc((size_t)NN * 4);
  float* sumlog = (float*)alloc(256);
  float* s1     = (float*)alloc((size_t)NN * FD * 4);
  float* s2     = (float*)alloc((size_t)NN * FD * 4);
  float* mx     = (float*)alloc((size_t)NN * FD * 4);
  float* mn     = (float*)alloc((size_t)NN * FD * 4);
  float* agg    = (float*)alloc((size_t)NN * 256 * 4);
  float* hA     = (float*)alloc((size_t)NN * FD * 4);
  float* hB     = (float*)alloc((size_t)NN * FD * 4);
  _Float16* preT16  = (_Float16*)alloc((size_t)NL * 64 * 128 * 2);
  _Float16* postT16 = (_Float16*)alloc((size_t)NL * 64 * 832 * 2);
  _Float16* linT16  = (_Float16*)alloc((size_t)NL * 64 * 64 * 2);

  // degree + scalers (once)
  k_zero<<<(NN + 255) / 256, 256, 0, stream>>>(deg, sumlog);
  k_deg<<<(NE + 255) / 256, 256, 0, stream>>>(dstp, deg);
  k_logdeg<<<(NN + 255) / 256, 256, 0, stream>>>(deg, logdeg, sumlog);
  k_scalers<<<(NN + 255) / 256, 256, 0, stream>>>(logdeg, sumlog, amp, att);

  // weight conversion fp32 -> f16 transposed (once, all layers)
  k_wcvt<<<(NL * 128 * 64 + 255) / 256, 256, 0, stream>>>(pre_w,  preT16,  128, 64, NL * 128 * 64);
  k_wcvt<<<(NL * 832 * 64 + 255) / 256, 256, 0, stream>>>(post_w, postT16, 832, 64, NL * 832 * 64);
  k_wcvt<<<(NL * 64 * 64 + 255) / 256, 256, 0, stream>>>(lin_w,  linT16,  64,  64, NL * 64 * 64);

  const int NODE_WAVES = 25;                                   // 3125 tiles = 25 * 125
  size_t nodeLds = (size_t)(64 * 832 + 64 * 64 + NODE_WAVES * 16 * 64) * 2; // 165,888 B
  hipFuncSetAttribute((const void*)k_node, hipFuncAttributeMaxDynamicSharedMemorySize, (int)nodeLds);

  const float* hin = x;
  float* houts[3] = { hA, hB, (float*)d_out };
  for (int l = 0; l < NL; ++l) {
    k_fill<<<(NN * FD + 255) / 256, 256, 0, stream>>>(s1, s2, mx, mn);
    k_edge<<<NE / 16 / 8, 256, 0, stream>>>(hin, srcp, dstp,
                                            preT16 + (size_t)l * 64 * 128, pre_b + l * 64,
                                            s1, s2, mx, mn);
    k_agg<<<(NN * FD + 255) / 256, 256, 0, stream>>>(deg, s1, s2, mx, mn, agg);
    k_node<<<NN / 16 / NODE_WAVES, NODE_WAVES * 32, nodeLds, stream>>>(
        hin, agg, amp, att,
        postT16 + (size_t)l * 64 * 832, post_b + l * 64,
        linT16 + (size_t)l * 64 * 64, lin_b + l * 64,
        houts[l]);
    hin = houts[l];
  }
}
